// MinGRU_46634754900275
// MI455X (gfx1250) — matrix operand order repaired and checked by reading, compiled, not run on hardware
//
#include <hip/hip_runtime.h>
#include <hip/hip_bf16.h>
#include <math.h>

// ---------------------------------------------------------------------------
// MinGRU forward for MI455X (gfx1250, wave32, WMMA)
//   x:(8,4096,512) fp32, W:(512,1024) fp32, b:(1024) fp32
//   out:(8,4096,512) fp32  ++  last_hidden:(8,1,512) fp32
// Pipeline:
//   k0: W -> bf16 hi/lo transposed planes Wt[n][k]                  (2 MB, L2)
//   k1: y = x@W + b via v_wmma_f32_16x16x32_bf16, bf16x3 split
//       - 1 wave = 16 rows x 128 cols (8 n-tiles) -> 24 WMMA / k-step
//       - x loads NT (streamed once), y stores RT (128 MB, L2-resident,
//         re-read twice by the scan passes; global L2 = 192 MB)
//   k2: per-chunk log-space scan totals (16 chunks of 256 over T)
//   k3: exclusive combine of chunk carries per channel
//   k4: local rescan fused with prefix -> out = exp(log_h) (NT stores)
// ---------------------------------------------------------------------------

typedef __attribute__((ext_vector_type(16))) __bf16 v16bf;
typedef __attribute__((ext_vector_type(8)))  __bf16 v8bf;
typedef __attribute__((ext_vector_type(8)))  float  v8f;
typedef __attribute__((ext_vector_type(4)))  float  v4f;

#define N_     8
#define T_     4096
#define C_     512
#define H_     512
#define NCOL_  1024          // 2H
#define M_     32768         // N*T
#define NCHUNK 16
#define CHLEN  256           // T / NCHUNK
#define NCHAN  4096          // N*H

// ---------------- math helpers (stable) ----------------
__device__ __forceinline__ float softplusf_(float x) {
  // log(1 + e^x) = max(x,0) + log1p(e^{-|x|})
  return fmaxf(x, 0.0f) + log1pf(expf(-fabsf(x)));
}
__device__ __forceinline__ float log_gf_(float x) {
  return (x >= 0.0f) ? logf(x + 0.5f) : -softplusf_(-x);
}
__device__ __forceinline__ float logaddexpf_(float p, float q) {
  float m = fmaxf(p, q);
  if (m == -INFINITY) return -INFINITY;
  return m + log1pf(expf(fminf(p, q) - m));
}

// ---------------- k0: W (C x 2H) fp32 -> Wt hi/lo bf16 (2H x C) ----------------
__global__ void __launch_bounds__(256)
mingru_wsplit(const float* __restrict__ W, __bf16* __restrict__ Whi,
              __bf16* __restrict__ Wlo) {
  int i = blockIdx.x * 256 + threadIdx.x;      // 0 .. 2H*C-1
  int n = i >> 9;                              // col of W  (0..1023)
  int k = i & (C_ - 1);                        // row of W  (0..511)
  float w = W[(size_t)k * NCOL_ + n];
  __bf16 h = (__bf16)w;
  Whi[(size_t)n * C_ + k] = h;
  Wlo[(size_t)n * C_ + k] = (__bf16)(w - (float)h);
}

// ---------------- k1: GEMM via WMMA, bf16x3 split, bias fused ----------------
// block = 256 threads = 8 waves; wave -> 16 rows x 128 cols (8 n-tiles)
// grid = (M/16) = 2048
__global__ void __launch_bounds__(256)
mingru_gemm(const float* __restrict__ x, const __bf16* __restrict__ Whi,
            const __bf16* __restrict__ Wlo, const float* __restrict__ bias,
            float* __restrict__ y) {
  const int lane = threadIdx.x & 31;
  const int wave = threadIdx.x >> 5;
  const int l16  = lane & 15;
  const int lh   = lane >> 4;                  // half-wave select
  const int m0   = blockIdx.x << 4;
  const int n0   = wave << 7;                  // 8 waves x 128 cols = 1024
  const float* xrow = x + (size_t)(m0 + l16) * C_;

  v8f acc[8];
#pragma unroll
  for (int t = 0; t < 8; ++t) {
    float bv = bias[n0 + (t << 4) + l16];
#pragma unroll
    for (int i = 0; i < 8; ++i) acc[t][i] = bv;
  }

  for (int k0 = 0; k0 < C_; k0 += 32) {
    // ---- A tile 16x32 bf16: lane=M; 16 contiguous K at k0+lh*16 ----
    const int ca = k0 + (lh << 4);
    v4f a0 = __builtin_nontemporal_load((const v4f*)(xrow + ca));
    v4f a1 = __builtin_nontemporal_load((const v4f*)(xrow + ca + 4));
    v4f a2 = __builtin_nontemporal_load((const v4f*)(xrow + ca + 8));
    v4f a3 = __builtin_nontemporal_load((const v4f*)(xrow + ca + 12));
    float af[16];
#pragma unroll
    for (int i = 0; i < 4; ++i) {
      af[i] = a0[i]; af[4 + i] = a1[i]; af[8 + i] = a2[i]; af[12 + i] = a3[i];
    }
    v16bf Ah, Al;
#pragma unroll
    for (int i = 0; i < 16; ++i) {
      __bf16 h = (__bf16)af[i];
      Ah[i] = h;
      Al[i] = (__bf16)(af[i] - (float)h);
    }
    // ---- B tiles 32x16 bf16: lane%16=N; 16 contiguous K at k0+lh*16 ----
    const int cb = k0 + (lh << 4);
#pragma unroll
    for (int t = 0; t < 8; ++t) {
      const size_t wb = (size_t)(n0 + (t << 4) + l16) * C_ + cb;
      v8bf h0 = *(const v8bf*)(Whi + wb);
      v8bf h1 = *(const v8bf*)(Whi + wb + 8);
      v8bf l0 = *(const v8bf*)(Wlo + wb);
      v8bf l1 = *(const v8bf*)(Wlo + wb + 8);
      v16bf Bh = __builtin_shufflevector(h0, h1, 0,1,2,3,4,5,6,7,8,9,10,11,12,13,14,15);
      v16bf Bl = __builtin_shufflevector(l0, l1, 0,1,2,3,4,5,6,7,8,9,10,11,12,13,14,15);
      acc[t] = __builtin_amdgcn_wmma_f32_16x16x32_bf16(false, Ah, false, Bh,
                                                       (short)0, acc[t], false, false);
      acc[t] = __builtin_amdgcn_wmma_f32_16x16x32_bf16(false, Ah, false, Bl,
                                                       (short)0, acc[t], false, false);
      acc[t] = __builtin_amdgcn_wmma_f32_16x16x32_bf16(false, Al, false, Bh,
                                                       (short)0, acc[t], false, false);
    }
  }
  // ---- C/D: VGPR r -> row m0 + 8*lh + r, col n0 + t*16 + l16 ----
#pragma unroll
  for (int t = 0; t < 8; ++t) {
    const int ncol = n0 + (t << 4) + l16;
#pragma unroll
    for (int r = 0; r < 8; ++r) {
      y[(size_t)(m0 + (lh << 3) + r) * NCOL_ + ncol] = acc[t][r];
    }
  }
}

// ---------------- k2: per-chunk scan totals ----------------
// grid = (NCHAN/256, NCHUNK), block 256; lane = channel (coalesced over h)
__global__ void __launch_bounds__(256)
mingru_scan_chunks(const float* __restrict__ y, float* __restrict__ cA,
                   float* __restrict__ cB) {
  const int c = blockIdx.x * 256 + threadIdx.x;       // channel n*H+h
  const int chunk = blockIdx.y;
  const int n = c >> 9, h = c & (H_ - 1);
  const size_t ybase = (size_t)n * T_ * NCOL_ + h;
  float A = 0.0f, B = -INFINITY;
  const int t0 = chunk * CHLEN;
#pragma unroll 4
  for (int t = t0; t < t0 + CHLEN; ++t) {
    float hid = y[ybase + (size_t)t * NCOL_];
    float g   = y[ybase + (size_t)t * NCOL_ + H_];
    float la = -softplusf_(g);
    float lb = -softplusf_(-g) + log_gf_(hid);
    B = logaddexpf_(B + la, lb);
    A += la;
  }
  cA[chunk * NCHAN + c] = A;
  cB[chunk * NCHAN + c] = B;
}

// ---------------- k3: exclusive combine across chunks ----------------
__global__ void __launch_bounds__(256)
mingru_scan_combine(const float* __restrict__ cA, const float* __restrict__ cB,
                    float* __restrict__ eA, float* __restrict__ eB) {
  const int c = blockIdx.x * 256 + threadIdx.x;       // channel
  float A = 0.0f, B = -INFINITY;
#pragma unroll
  for (int j = 0; j < NCHUNK; ++j) {
    eA[j * NCHAN + c] = A;
    eB[j * NCHAN + c] = B;
    float Ac = cA[j * NCHAN + c];
    float Bc = cB[j * NCHAN + c];
    B = logaddexpf_(B + Ac, Bc);
    A += Ac;
  }
}

// ---------------- k4: local rescan + prefix -> out, last_hidden ----------------
__global__ void __launch_bounds__(256)
mingru_scan_fix(const float* __restrict__ y, const float* __restrict__ eA,
                const float* __restrict__ eB, float* __restrict__ out,
                float* __restrict__ last) {
  const int c = blockIdx.x * 256 + threadIdx.x;
  const int chunk = blockIdx.y;
  const int n = c >> 9, h = c & (H_ - 1);
  const float Bp = eB[chunk * NCHAN + c];
  const size_t ybase = (size_t)n * T_ * NCOL_ + h;
  const size_t obase = (size_t)n * T_ * H_ + h;
  float Al = 0.0f, Bl = -INFINITY;
  const int t0 = chunk * CHLEN;
#pragma unroll 4
  for (int t = t0; t < t0 + CHLEN; ++t) {
    float hid = y[ybase + (size_t)t * NCOL_];
    float g   = y[ybase + (size_t)t * NCOL_ + H_];
    float la = -softplusf_(g);
    float lb = -softplusf_(-g) + log_gf_(hid);
    Bl = logaddexpf_(Bl + la, lb);
    Al += la;
    float lh  = logaddexpf_(Bp + Al, Bl);   // global log_h
    float val = expf(lh);
    __builtin_nontemporal_store(val, &out[obase + (size_t)t * H_]);
    if (t == T_ - 1) last[(size_t)n * H_ + h] = val;
  }
}

// ---------------------------------------------------------------------------
extern "C" void kernel_launch(void* const* d_in, const int* in_sizes, int n_in,
                              void* d_out, int out_size, void* d_ws, size_t ws_size,
                              hipStream_t stream) {
  (void)in_sizes; (void)n_in; (void)out_size; (void)ws_size;
  const float* x = (const float*)d_in[0];   // (8,4096,512)
  const float* W = (const float*)d_in[1];   // (512,1024)
  const float* b = (const float*)d_in[2];   // (1024)

  char* ws = (char*)d_ws;
  size_t off = 0;
  __bf16* Whi = (__bf16*)(ws + off); off += (size_t)NCOL_ * C_ * sizeof(__bf16); // 1 MB
  __bf16* Wlo = (__bf16*)(ws + off); off += (size_t)NCOL_ * C_ * sizeof(__bf16); // 1 MB
  float*  y   = (float*)(ws + off);  off += (size_t)M_ * NCOL_ * sizeof(float);  // 128 MB
  float*  cA  = (float*)(ws + off);  off += (size_t)NCHUNK * NCHAN * sizeof(float);
  float*  cB  = (float*)(ws + off);  off += (size_t)NCHUNK * NCHAN * sizeof(float);
  float*  eA  = (float*)(ws + off);  off += (size_t)NCHUNK * NCHAN * sizeof(float);
  float*  eB  = (float*)(ws + off);  off += (size_t)NCHUNK * NCHAN * sizeof(float);

  float* out  = (float*)d_out;                       // (8,4096,512)
  float* last = out + (size_t)N_ * T_ * H_;          // (8,1,512)

  mingru_wsplit<<<dim3((NCOL_ * C_) / 256), dim3(256), 0, stream>>>(W, Whi, Wlo);
  mingru_gemm<<<dim3(M_ / 16), dim3(256), 0, stream>>>(x, Whi, Wlo, b, y);
  mingru_scan_chunks<<<dim3(NCHAN / 256, NCHUNK), dim3(256), 0, stream>>>(y, cA, cB);
  mingru_scan_combine<<<dim3(NCHAN / 256), dim3(256), 0, stream>>>(cA, cB, eA, eB);
  mingru_scan_fix<<<dim3(NCHAN / 256, NCHUNK), dim3(256), 0, stream>>>(y, eA, eB, out, last);
}
